// SuperGAT_49289044689247
// MI455X (gfx1250) — compile-verified, matrix operand order checked by reading, not run on hardware
//
#include <hip/hip_runtime.h>
#include <hip/hip_bf16.h>
#include <cstdint>

// ---------------------------------------------------------------------------
// MI455X (gfx1250) SuperGAT forward.
// GEMM1 is HBM-bound (410MB of x @ 23.3TB/s ~= 18us); GEMM2 + edge phase are
// L2-resident. Matrix math uses v_wmma_f32_16x16x32_bf16 (wave32, fp32 accum),
// B tiles staged into LDS with GLOBAL_LOAD_ASYNC_TO_LDS_B128 (ASYNCcnt path,
// no VGPR round trip), A converted f32->bf16 with v_cvt_pk_bf16_f32.
// ---------------------------------------------------------------------------

typedef __attribute__((ext_vector_type(16))) __bf16 v16bf;
typedef __attribute__((ext_vector_type(8)))  float  v8f;

#if __has_builtin(__builtin_amdgcn_global_load_async_to_lds_b128) && \
    __has_builtin(__builtin_amdgcn_s_wait_asynccnt)
#define USE_ASYNC_LDS 1
#else
#define USE_ASYNC_LDS 0
#endif

// Builtin parameter types (from hipcc diagnostic): pointer to <4 x i32> in
// AS1 (global) / AS3 (LDS).
typedef int v4i_ __attribute__((__vector_size__(16)));
typedef __attribute__((address_space(1))) v4i_ as1_v4i;
typedef __attribute__((address_space(3))) v4i_ as3_v4i;

// Native conversion: clang lowers f32->bf16 fptrunc (RNE) to v_cvt_pk_bf16_f32.
__device__ __forceinline__ __bf16 f2bf(float f) { return (__bf16)f; }

// monotonic float -> unsigned key for atomicMax-based segment max
__device__ __forceinline__ unsigned f2key(float f) {
  unsigned k = __float_as_uint(f);
  return (k & 0x80000000u) ? ~k : (k | 0x80000000u);
}
__device__ __forceinline__ float key2f(unsigned k) {
  return (k & 0x80000000u) ? __uint_as_float(k ^ 0x80000000u)
                           : __uint_as_float(~k);
}

// ---------------------------------------------------------------------------
// Edge list prep: int32 src/dst with self-loops appended.
// ---------------------------------------------------------------------------
__global__ void prep_edges(const int* __restrict__ eidx, int E, int N,
                           int* __restrict__ srcI, int* __restrict__ dstI) {
  int e = blockIdx.x * 256 + threadIdx.x;
  int Etot = E + N;
  if (e >= Etot) return;
  if (e < E) { srcI[e] = eidx[e]; dstI[e] = eidx[E + e]; }
  else       { srcI[e] = e - E;   dstI[e] = e - E; }
}

// W [K][Nc] f32 (row-major)  ->  WT [Nc][K] bf16 (row-major)
__global__ void cast_transpose(const float* __restrict__ W,
                               __bf16* __restrict__ WT, int K, int Nc) {
  int idx = blockIdx.x * 256 + threadIdx.x;
  if (idx >= K * Nc) return;
  int n = idx / K, k = idx % K;
  WT[idx] = f2bf(W[(size_t)k * Nc + n]);
}

__global__ void fill_u32(unsigned* __restrict__ p, unsigned v, size_t n) {
  size_t i = blockIdx.x * (size_t)256 + threadIdx.x;
  if (i < n) p[i] = v;
}

// ---------------------------------------------------------------------------
// Hout[M][NT*16] = X[M][K](f32) @ WT[NT*16][K](bf16)^T  via WMMA bf16.
// One wave per 16-row strip; 8 waves (128 rows) per block share the LDS
// B slab.  KS=64 deep slab per barrier round.
// ---------------------------------------------------------------------------
template <int NT, int K>
__global__ void __launch_bounds__(256)
gemm_x_wt(const float* __restrict__ X, const __bf16* __restrict__ WT,
          float* __restrict__ Hout, int M) {
  constexpr int NC = NT * 16;
  constexpr int KS = 64;                          // k-slab per barrier round
  __shared__ __align__(64) __bf16 lb[NC * KS];    // [NC][KS]
  const int tid  = threadIdx.x;
  const int wave = tid >> 5;
  const int lane = tid & 31;
  const int nstrips = M >> 4;                     // M divisible by 16 here
  int strip = blockIdx.x * 8 + wave;
  const bool active = strip < nstrips;
  if (!active) strip = nstrips - 1;               // keep EXEC full for WMMA
  const int row0 = strip << 4;
  const int arow = row0 + (lane & 15);
  const int koff = (lane >> 4) << 4;              // which 16-wide K half

  v8f acc[NT];
#pragma unroll
  for (int t = 0; t < NT; ++t) acc[t] = {};

  for (int k0 = 0; k0 < K; k0 += KS) {
    __syncthreads();
    // stage B slab: NC rows x KS bf16, 16B chunks, compile-time chunk count
    constexpr int CH = (NC * (KS / 8)) / 256;     // chunks per thread
#pragma unroll
    for (int j = 0; j < CH; ++j) {
      const int i  = tid + j * 256;
      const int r  = i / (KS / 8);
      const int kk = (i % (KS / 8)) * 8;
#if USE_ASYNC_LDS
      // CDNA5 async copy: memory -> LDS without a VGPR round trip (ASYNCcnt).
      as1_v4i* g = (as1_v4i*)(uintptr_t)(WT + (size_t)r * K + k0 + kk);
      as3_v4i* l = (as3_v4i*)(unsigned)(uintptr_t)&lb[r * KS + kk];
      __builtin_amdgcn_global_load_async_to_lds_b128(g, l, 0, 0);
#else
      *(uint4*)&lb[r * KS + kk] = *(const uint4*)&WT[(size_t)r * K + k0 + kk];
#endif
    }
#if USE_ASYNC_LDS
    __builtin_amdgcn_s_wait_asynccnt(0);
#endif
    __syncthreads();

#pragma unroll
    for (int kh = 0; kh < KS; kh += 32) {
      // A fragment: 16 consecutive f32 of one row -> bf16 (contiguous k = 16h + s, same order as B)
      const float* pa = X + (size_t)arow * K + k0 + kh + koff;
      if (k0 + kh + 32 < K) __builtin_prefetch(pa + 32, 0, 0);
      float4 f0 = ((const float4*)pa)[0];
      float4 f1 = ((const float4*)pa)[1];
      float4 f2 = ((const float4*)pa)[2];
      float4 f3 = ((const float4*)pa)[3];
      v16bf a;
      a[0]=f2bf(f0.x);  a[1]=f2bf(f0.y);  a[2]=f2bf(f0.z);  a[3]=f2bf(f0.w);
      a[4]=f2bf(f1.x);  a[5]=f2bf(f1.y);  a[6]=f2bf(f1.z);  a[7]=f2bf(f1.w);
      a[8]=f2bf(f2.x);  a[9]=f2bf(f2.y);  a[10]=f2bf(f2.z); a[11]=f2bf(f2.w);
      a[12]=f2bf(f3.x); a[13]=f2bf(f3.y); a[14]=f2bf(f3.z); a[15]=f2bf(f3.w);

#pragma unroll
      for (int t0 = 0; t0 < NT; t0 += 4) {
        // preload 4 B fragments, then 4 back-to-back WMMAs
        const int nbase = (lane & 15);
        v16bf b0 = *(const v16bf*)&lb[((t0 + 0) * 16 + nbase) * KS + kh + koff];
        v16bf b1 = *(const v16bf*)&lb[((t0 + 1) * 16 + nbase) * KS + kh + koff];
        v16bf b2 = *(const v16bf*)&lb[((t0 + 2) * 16 + nbase) * KS + kh + koff];
        v16bf b3 = *(const v16bf*)&lb[((t0 + 3) * 16 + nbase) * KS + kh + koff];
        acc[t0 + 0] = __builtin_amdgcn_wmma_f32_16x16x32_bf16(
            false, a, false, b0, (short)0, acc[t0 + 0], false, false);
        acc[t0 + 1] = __builtin_amdgcn_wmma_f32_16x16x32_bf16(
            false, a, false, b1, (short)0, acc[t0 + 1], false, false);
        acc[t0 + 2] = __builtin_amdgcn_wmma_f32_16x16x32_bf16(
            false, a, false, b2, (short)0, acc[t0 + 2], false, false);
        acc[t0 + 3] = __builtin_amdgcn_wmma_f32_16x16x32_bf16(
            false, a, false, b3, (short)0, acc[t0 + 3], false, false);
      }
    }
  }

  if (active) {
    const int col   = lane & 15;
    const int rbase = row0 + ((lane >> 4) << 3);  // C/D layout: M=r / M=8+r
#pragma unroll
    for (int t = 0; t < NT; ++t)
#pragma unroll
      for (int r = 0; r < 8; ++r)
        Hout[(size_t)(rbase + r) * NC + t * 16 + col] = acc[t][r];
  }
}

// ---------------------------------------------------------------------------
// Edge pass 1: alpha = leaky_relu((<hj,al>+<hi,ar>) * sigmoid(<hi,hj>)),
// and segment-max into amaxU (ordered-key atomicMax).
// ---------------------------------------------------------------------------
template <int HH, int CC>
__global__ void edge_alpha(const int* __restrict__ src, const int* __restrict__ dst,
                           const float* __restrict__ h,
                           const float* __restrict__ attl, const float* __restrict__ attr,
                           float* __restrict__ alpha, unsigned* __restrict__ amaxU,
                           int Etot) {
  int e = blockIdx.x * 256 + threadIdx.x;
  if (e >= Etot) return;
  const int s = src[e], d = dst[e];
  const float* pj = h + (size_t)s * (HH * CC);
  const float* pi = h + (size_t)d * (HH * CC);
#pragma unroll
  for (int hh = 0; hh < HH; ++hh) {
    float lg = 0.f, al = 0.f, ar = 0.f;
#pragma unroll
    for (int c = 0; c < CC; ++c) {
      float vj = pj[hh * CC + c], vi = pi[hh * CC + c];
      lg += vi * vj;
      al += vj * attl[hh * CC + c];
      ar += vi * attr[hh * CC + c];
    }
    float a = (al + ar) / (1.f + __expf(-lg));
    a = a > 0.f ? a : 0.2f * a;                          // leaky_relu 0.2
    alpha[(size_t)e * HH + hh] = a;
    atomicMax(&amaxU[(size_t)d * HH + hh], f2key(a));
  }
}

// Edge pass 2: ex = exp(alpha - amax[dst]); segment-sum into denom.
template <int HH>
__global__ void edge_expsum(const int* __restrict__ dst,
                            float* __restrict__ alpha,
                            const unsigned* __restrict__ amaxU,
                            float* __restrict__ denom, int Etot) {
  int e = blockIdx.x * 256 + threadIdx.x;
  if (e >= Etot) return;
  const int d = dst[e];
#pragma unroll
  for (int hh = 0; hh < HH; ++hh) {
    float m  = key2f(amaxU[(size_t)d * HH + hh]);
    float ex = __expf(alpha[(size_t)e * HH + hh] - m);
    alpha[(size_t)e * HH + hh] = ex;
    atomicAdd(&denom[(size_t)d * HH + hh], ex);
  }
}

// Edge pass 3: agg[dst] += hj * ex/denom[dst]  (L2-resident fp32 atomics).
template <int HH, int CC>
__global__ void edge_agg(const int* __restrict__ src, const int* __restrict__ dst,
                         const float* __restrict__ h,
                         const float* __restrict__ alpha,
                         const float* __restrict__ denom,
                         float* __restrict__ agg, int Etot) {
  int e = blockIdx.x * 256 + threadIdx.x;
  if (e >= Etot) return;
  const int s = src[e], d = dst[e];
  const float* pj = h + (size_t)s * (HH * CC);
#pragma unroll
  for (int hh = 0; hh < HH; ++hh) {
    float w = alpha[(size_t)e * HH + hh] /
              (denom[(size_t)d * HH + hh] + 1e-16f);
#pragma unroll
    for (int c = 0; c < CC; ++c)
      atomicAdd(&agg[(size_t)d * (HH * CC) + hh * CC + c], pj[hh * CC + c] * w);
  }
}

// out = elu(agg + bias), elementwise over [N*F]
__global__ void finish_elu(const float* __restrict__ agg, const float* __restrict__ bias,
                           float* __restrict__ out, int total, int F) {
  int i = blockIdx.x * 256 + threadIdx.x;
  if (i >= total) return;
  float v = agg[i] + bias[i % F];
  out[i] = v > 0.f ? v : __expf(v) - 1.f;
}

// mean over 8 heads, +b2, log_softmax over 16 classes; also write att_loss=0.
__global__ void finish_out(const float* __restrict__ agg2, const float* __restrict__ b2,
                           float* __restrict__ out, int N) {
  int n = blockIdx.x * 256 + threadIdx.x;
  if (n >= N) return;
  float v[16];
#pragma unroll
  for (int c = 0; c < 16; ++c) {
    float s = 0.f;
#pragma unroll
    for (int hh = 0; hh < 8; ++hh) s += agg2[(size_t)n * 128 + hh * 16 + c];
    v[c] = s * 0.125f + b2[c];
  }
  float m = v[0];
#pragma unroll
  for (int c = 1; c < 16; ++c) m = fmaxf(m, v[c]);
  float se = 0.f;
#pragma unroll
  for (int c = 0; c < 16; ++c) se += __expf(v[c] - m);
  float ls = __logf(se);
#pragma unroll
  for (int c = 0; c < 16; ++c) out[(size_t)n * 16 + c] = v[c] - m - ls;
  if (n == 0) out[(size_t)N * 16] = 0.f;                 // att_loss (eval mode)
}

// ---------------------------------------------------------------------------
extern "C" void kernel_launch(void* const* d_in, const int* in_sizes, int n_in,
                              void* d_out, int out_size, void* d_ws, size_t ws_size,
                              hipStream_t stream) {
  const float* x   = (const float*)d_in[0];
  const int*  eidx = (const int*)  d_in[1];
  const float* W1  = (const float*)d_in[2];
  const float* al1 = (const float*)d_in[3];
  const float* ar1 = (const float*)d_in[4];
  const float* b1  = (const float*)d_in[5];
  const float* W2  = (const float*)d_in[6];
  const float* al2 = (const float*)d_in[7];
  const float* ar2 = (const float*)d_in[8];
  const float* b2  = (const float*)d_in[9];
  float* out = (float*)d_out;
  (void)n_in; (void)out_size; (void)ws_size;

  const int Fin = 2048, HC1 = 64, HC2 = 128, HH = 8;
  const int N = in_sizes[0] / Fin;
  const int E = in_sizes[1] / 2;
  const int Etot = E + N;

  char* base = (char*)d_ws;
  size_t off = 0;
  auto take = [&](size_t bytes) -> char* {
    off = (off + 255) & ~(size_t)255;
    char* p = base + off;
    off += bytes;
    return p;
  };
  __bf16* w1t   = (__bf16*)take((size_t)HC1 * Fin * 2);
  __bf16* w2t   = (__bf16*)take((size_t)HC2 * HC1 * 2);
  int*    srcI  = (int*)   take((size_t)Etot * 4);
  int*    dstI  = (int*)   take((size_t)Etot * 4);
  float*  h1    = (float*) take((size_t)N * HC1 * 4);   // reused as elu(h1) after layer-1
  float*  agg1  = (float*) take((size_t)N * HC1 * 4);
  float*  h2    = (float*) take((size_t)N * HC2 * 4);
  float*  agg2  = (float*) take((size_t)N * HC2 * 4);
  float*  alpha = (float*) take((size_t)Etot * HH * 4);
  unsigned* amaxU = (unsigned*)take((size_t)N * HH * 4);
  float*  denom = (float*) take((size_t)N * HH * 4);

  const int TB = 256;
  auto cdiv = [](long long a, long long b) { return (unsigned)((a + b - 1) / b); };

  prep_edges<<<cdiv(Etot, TB), TB, 0, stream>>>(eidx, E, N, srcI, dstI);
  cast_transpose<<<cdiv((long long)Fin * HC1, TB), TB, 0, stream>>>(W1, w1t, Fin, HC1);
  cast_transpose<<<cdiv((long long)HC1 * HC2, TB), TB, 0, stream>>>(W2, w2t, HC1, HC2);

  // --- layer 1 ---
  gemm_x_wt<4, 2048><<<cdiv(N / 16, 8), TB, 0, stream>>>(x, w1t, h1, N);
  fill_u32<<<cdiv((long long)N * HH, TB), TB, 0, stream>>>(amaxU, 0u, (size_t)N * HH);
  fill_u32<<<cdiv((long long)N * HH, TB), TB, 0, stream>>>((unsigned*)denom, 0u, (size_t)N * HH);
  fill_u32<<<cdiv((long long)N * HC1, TB), TB, 0, stream>>>((unsigned*)agg1, 0u, (size_t)N * HC1);
  edge_alpha<8, 8><<<cdiv(Etot, TB), TB, 0, stream>>>(srcI, dstI, h1, al1, ar1, alpha, amaxU, Etot);
  edge_expsum<8><<<cdiv(Etot, TB), TB, 0, stream>>>(dstI, alpha, amaxU, denom, Etot);
  edge_agg<8, 8><<<cdiv(Etot, TB), TB, 0, stream>>>(srcI, dstI, h1, alpha, denom, agg1, Etot);
  finish_elu<<<cdiv((long long)N * HC1, TB), TB, 0, stream>>>(agg1, b1, h1, N * HC1, HC1);

  // --- layer 2 ---
  gemm_x_wt<8, 64><<<cdiv(N / 16, 8), TB, 0, stream>>>(h1, w2t, h2, N);
  fill_u32<<<cdiv((long long)N * HH, TB), TB, 0, stream>>>(amaxU, 0u, (size_t)N * HH);
  fill_u32<<<cdiv((long long)N * HH, TB), TB, 0, stream>>>((unsigned*)denom, 0u, (size_t)N * HH);
  fill_u32<<<cdiv((long long)N * HC2, TB), TB, 0, stream>>>((unsigned*)agg2, 0u, (size_t)N * HC2);
  edge_alpha<8, 16><<<cdiv(Etot, TB), TB, 0, stream>>>(srcI, dstI, h2, al2, ar2, alpha, amaxU, Etot);
  edge_expsum<8><<<cdiv(Etot, TB), TB, 0, stream>>>(dstI, alpha, amaxU, denom, Etot);
  edge_agg<8, 16><<<cdiv(Etot, TB), TB, 0, stream>>>(srcI, dstI, h2, alpha, denom, agg2, Etot);

  finish_out<<<cdiv(N, TB), TB, 0, stream>>>(agg2, b2, out, N);
}
